// MultiHeadAttention_67645734912496
// MI455X (gfx1250) — matrix operand order repaired and checked by reading, not compiled, not run on hardware
//
#include <hip/hip_runtime.h>

typedef __attribute__((ext_vector_type(16))) __bf16 v16bf;
typedef __attribute__((ext_vector_type(8)))  __bf16 v8bf;
typedef __attribute__((ext_vector_type(8)))  float  v8f;
typedef __attribute__((ext_vector_type(4)))  int    v4i;

typedef __attribute__((address_space(1))) v4i* gv4i_p;   // global int4*
typedef __attribute__((address_space(3))) v4i* lv4i_p;   // LDS int4*

#define WMMA_BF16(A, Bm, Cm) \
  __builtin_amdgcn_wmma_f32_16x16x32_bf16(false, (A), false, (Bm), (short)0, (Cm), false, false)

#if __has_builtin(__builtin_amdgcn_global_load_async_to_lds_b128) && \
    __has_builtin(__builtin_amdgcn_s_wait_asynccnt)
#define USE_ASYNC_LDS 1
#endif

constexpr int B_  = 2;
constexpr int L_  = 4096;
constexpr int C_  = 512;
constexpr int H_  = 8;
constexpr int DH_ = 64;
constexpr int M_  = B_ * L_;   // 8192

// copy 8 bf16 (16 bytes) global -> LDS
__device__ __forceinline__ void cp16(const __bf16* g, __bf16* l) {
#ifdef USE_ASYNC_LDS
  __builtin_amdgcn_global_load_async_to_lds_b128((gv4i_p)g, (lv4i_p)l, 0, 0);
#else
  *(v8bf*)l = *(const v8bf*)g;
#endif
}

__device__ __forceinline__ void async_wait() {
#ifdef USE_ASYNC_LDS
  __builtin_amdgcn_s_wait_asynccnt(0);
#endif
}

// ---------------------------------------------------------------- converts
__global__ void k_cvt(const float* __restrict__ in, __bf16* __restrict__ out, int n) {
  int i = blockIdx.x * blockDim.x + threadIdx.x;
  if (i < n) out[i] = (__bf16)in[i];
}

// Wt[n*C + k] = (bf16) W[k*C + n]
__global__ void k_cvtT(const float* __restrict__ W, __bf16* __restrict__ Wt) {
  int i = blockIdx.x * blockDim.x + threadIdx.x;   // i = n*C + k
  int n = i >> 9;
  int k = i & (C_ - 1);
  Wt[i] = (__bf16)W[k * C_ + n];
}

// ---------------------------------------------------------------- QKV GEMM
// Y[m,n] = sum_k X[m,k] * Wt[n,k]; each wave: 64x64 tile (4x4 WMMA accs).
// mode 1: store [B,H,L,DH] (Q,K);  mode 2: store [B,H,DH,L] (V transposed)
__global__ __launch_bounds__(128) void k_gemm_qkv(const __bf16* __restrict__ X,
                                                  const __bf16* __restrict__ Wt,
                                                  __bf16* __restrict__ Out, int mode) {
  const int lane = threadIdx.x & 31;
  const int wave = threadIdx.x >> 5;
  const int half = lane >> 4;
  const int ln   = lane & 15;
  const int rowBase = blockIdx.x * 256 + wave * 64;
  const int colBase = blockIdx.y * 64;

  v8f acc[4][4];
#pragma unroll
  for (int a = 0; a < 4; ++a)
#pragma unroll
    for (int b = 0; b < 4; ++b)
#pragma unroll
      for (int e = 0; e < 8; ++e) acc[a][b][e] = 0.f;

  union F16 { v16bf v; v8bf h[2]; };

  for (int k0 = 0; k0 < C_; k0 += 32) {
    F16 af[4], bf[4];
#pragma unroll
    for (int mt = 0; mt < 4; ++mt) {
      const __bf16* p = X + (size_t)(rowBase + mt * 16 + ln) * C_ + k0 + half * 16;
      af[mt].h[0] = *(const v8bf*)p;
      af[mt].h[1] = *(const v8bf*)(p + 8);
    }
#pragma unroll
    for (int nt = 0; nt < 4; ++nt) {
      const __bf16* p = Wt + (size_t)(colBase + nt * 16 + ln) * C_ + k0 + half * 16;
      bf[nt].v = *(const v16bf*)p;
    }
#pragma unroll
    for (int mt = 0; mt < 4; ++mt)
#pragma unroll
      for (int nt = 0; nt < 4; ++nt)
        acc[mt][nt] = WMMA_BF16(af[mt].v, bf[nt].v, acc[mt][nt]);
  }

#pragma unroll
  for (int mt = 0; mt < 4; ++mt)
#pragma unroll
    for (int nt = 0; nt < 4; ++nt) {
      const int n  = colBase + nt * 16 + ln;
      const int h  = n >> 6;
      const int dh = n & (DH_ - 1);
#pragma unroll
      for (int i = 0; i < 8; ++i) {
        const int m  = rowBase + mt * 16 + half * 8 + i;
        const int bb = m >> 12;
        const int l  = m & (L_ - 1);
        const __bf16 vv = (__bf16)acc[mt][nt][i];
        if (mode == 1)
          Out[(size_t)((bb * H_ + h) * L_ + l) * DH_ + dh] = vv;
        else
          Out[((size_t)(bb * H_ + h) * DH_ + dh) * L_ + l] = vv;
      }
    }
}

// ---------------------------------------------------------------- attention
// WG = 4 waves, each wave owns 32 q rows (two 16-row tiles) of one (b,h).
// K/V blocks of 32 keys staged in LDS (double-buffered, async prefetch),
// shared by all 4 waves. Flash softmax per q tile.
__global__ __launch_bounds__(128) void k_attn(const __bf16* __restrict__ Q,
                                              const __bf16* __restrict__ K,
                                              const __bf16* __restrict__ Vt,
                                              __bf16* __restrict__ O) {
  __shared__ __attribute__((aligned(16))) __bf16 Ks[2][32 * DH_]; // [buf][kpos*64+dh]
  __shared__ __attribute__((aligned(16))) __bf16 Vs[2][DH_ * 32]; // [buf][dh*32+kpos]

  const int tid  = threadIdx.x;
  const int lane = tid & 31;
  const int wave = tid >> 5;
  const int half = lane >> 4;
  const int ln   = lane & 15;
  const int bh   = blockIdx.y;                   // b*H + h
  const int qb0  = blockIdx.x * 128 + wave * 32; // this wave: q rows qb0..qb0+31

  const __bf16* Qh = Q  + (size_t)bh * L_ * DH_;
  const __bf16* Kh = K  + (size_t)bh * L_ * DH_;
  const __bf16* Vh = Vt + (size_t)bh * DH_ * L_;

  union F16 { v16bf v; v8bf h[2]; };

  // B-fragments of Q^T per q tile qt, dh chunk c
  F16 bq[2][2];
#pragma unroll
  for (int qt = 0; qt < 2; ++qt) {
    const __bf16* p = Qh + (size_t)(qb0 + qt * 16 + ln) * DH_ + half * 16;
    bq[qt][0].v = *(const v16bf*)p;
    bq[qt][1].v = *(const v16bf*)(p + 32);
  }

  v8f o[2][4];
#pragma unroll
  for (int qt = 0; qt < 2; ++qt)
#pragma unroll
    for (int t = 0; t < 4; ++t)
#pragma unroll
      for (int i = 0; i < 8; ++i) o[qt][t][i] = 0.f;

  v8f zero;
#pragma unroll
  for (int i = 0; i < 8; ++i) zero[i] = 0.f;

  float m_run[2] = {-3.0e38f, -3.0e38f};
  float l_run[2] = {0.f, 0.f};
  const float scale = 0.125f;   // 1/sqrt(DH)

  // cooperative stage of one 32-key block into LDS buffer `buf`
  auto stage = [&](int buf, int kb) {
    // K block is contiguous: 2048 bf16 starting at Kh + kb*64; 16 bf16/thread
    {
      const __bf16* g = Kh + (size_t)kb * DH_ + tid * 16;
      __bf16* l = &Ks[buf][tid * 16];
      cp16(g, l); cp16(g + 8, l + 8);
    }
    // V^T block: 64 rows (dh), 32 bf16 each, stride L; 2 threads per row
    {
      const int dh = tid >> 1, ho = (tid & 1) * 16;
      const __bf16* g = Vh + (size_t)dh * L_ + kb + ho;
      __bf16* l = &Vs[buf][dh * 32 + ho];
      cp16(g, l); cp16(g + 8, l + 8);
    }
  };

  stage(0, 0);   // prologue prefetch

  for (int kb = 0; kb < L_; kb += 32) {
    const int cbuf = (kb >> 5) & 1;
    async_wait();        // my prefetch for cbuf complete
    __syncthreads();     // all threads' writes visible; prior reads of other buf done
    if (kb + 32 < L_) stage(cbuf ^ 1, kb + 32);

    // A-fragments of K (2 kpos sub-tiles x 2 dh chunks) from LDS
    F16 ak[2][2];
#pragma unroll
    for (int tt = 0; tt < 2; ++tt)
#pragma unroll
      for (int c = 0; c < 2; ++c) {
        const __bf16* p = &Ks[cbuf][(tt * 16 + ln) * DH_ + c * 32 + half * 16];
        ak[tt][c].h[0] = *(const v8bf*)p;
        ak[tt][c].h[1] = *(const v8bf*)(p + 8);
      }
    // A-fragments of V^T (4 dh tiles) from LDS
    F16 av[4];
#pragma unroll
    for (int t = 0; t < 4; ++t) {
      const __bf16* p = &Vs[cbuf][(t * 16 + ln) * 32 + half * 16];
      av[t].h[0] = *(const v8bf*)p;
      av[t].h[1] = *(const v8bf*)(p + 8);
    }

#pragma unroll
    for (int qt = 0; qt < 2; ++qt) {
      v8f s0 = WMMA_BF16(ak[0][0].v, bq[qt][0].v, zero);
      s0     = WMMA_BF16(ak[0][1].v, bq[qt][1].v, s0);
      v8f s1 = WMMA_BF16(ak[1][0].v, bq[qt][0].v, zero);
      s1     = WMMA_BF16(ak[1][1].v, bq[qt][1].v, s1);

      // online softmax over this 32-key block (lane = q column)
      float mloc = -3.0e38f;
#pragma unroll
      for (int i = 0; i < 8; ++i) {
        s0[i] *= scale; s1[i] *= scale;
        mloc = fmaxf(mloc, fmaxf(s0[i], s1[i]));
      }
      mloc = fmaxf(mloc, __shfl_xor(mloc, 16));
      const float m_new = fmaxf(m_run[qt], mloc);
      const float alpha = __expf(m_run[qt] - m_new);

      float p0[8], p1[8];
      float lsum = 0.f;
#pragma unroll
      for (int i = 0; i < 8; ++i) {
        p0[i] = __expf(s0[i] - m_new);
        p1[i] = __expf(s1[i] - m_new);
        lsum += p0[i] + p1[i];
      }
      lsum += __shfl_xor(lsum, 16);
      l_run[qt] = l_run[qt] * alpha + lsum;
      m_run[qt] = m_new;

#pragma unroll
      for (int t = 0; t < 4; ++t)
#pragma unroll
        for (int i = 0; i < 8; ++i) o[qt][t][i] *= alpha;

      // build P^T B-fragment (32 kpos x 16 q) via cross-half shuffles
      v16bf pf;
#pragma unroll
      for (int j = 0; j < 8; ++j) {
        const float c0 = __shfl_xor(p0[j], 16);
        const float c1 = __shfl_xor(p1[j], 16);
        pf[j]     = (__bf16)(half ? c1    : p0[j]);
        pf[j + 8] = (__bf16)(half ? p1[j] : c0);
      }

      // O^T += V^T * P^T
#pragma unroll
      for (int t = 0; t < 4; ++t)
        o[qt][t] = WMMA_BF16(av[t].v, pf, o[qt][t]);
    }
  }

  // epilogue: normalize, store att in [B, L, C] (bf16), 16B per store
  const int b = bh >> 3, h = bh & 7;
#pragma unroll
  for (int qt = 0; qt < 2; ++qt) {
    const float inv = 1.f / l_run[qt];
#pragma unroll
    for (int t = 0; t < 4; ++t) {
      v8bf st;
#pragma unroll
      for (int i = 0; i < 8; ++i) st[i] = (__bf16)(o[qt][t][i] * inv);
      __bf16* p = O + (size_t)(b * L_ + qb0 + qt * 16 + ln) * C_ +
                  h * DH_ + t * 16 + half * 8;
      *(v8bf*)p = st;
    }
  }
}

// ---------------------------------------------------------------- out GEMM
// out[m,n] = sum_k att[m,k]*Wot[n,k] + bo[n]   (f32 output)
__global__ __launch_bounds__(128) void k_gemm_out(const __bf16* __restrict__ X,
                                                  const __bf16* __restrict__ Wt,
                                                  const float* __restrict__ bias,
                                                  float* __restrict__ Out) {
  const int lane = threadIdx.x & 31;
  const int wave = threadIdx.x >> 5;
  const int half = lane >> 4;
  const int ln   = lane & 15;
  const int rowBase = blockIdx.x * 256 + wave * 64;
  const int colBase = blockIdx.y * 64;

  v8f acc[4][4];
#pragma unroll
  for (int a = 0; a < 4; ++a)
#pragma unroll
    for (int b = 0; b < 4; ++b)
#pragma unroll
      for (int e = 0; e < 8; ++e) acc[a][b][e] = 0.f;

  union F16 { v16bf v; v8bf h[2]; };

  for (int k0 = 0; k0 < C_; k0 += 32) {
    F16 af[4], bf[4];
#pragma unroll
    for (int mt = 0; mt < 4; ++mt) {
      const __bf16* p = X + (size_t)(rowBase + mt * 16 + ln) * C_ + k0 + half * 16;
      af[mt].h[0] = *(const v8bf*)p;
      af[mt].h[1] = *(const v8bf*)(p + 8);
    }
#pragma unroll
    for (int nt = 0; nt < 4; ++nt) {
      const __bf16* p = Wt + (size_t)(colBase + nt * 16 + ln) * C_ + k0 + half * 16;
      bf[nt].v = *(const v16bf*)p;
    }
#pragma unroll
    for (int mt = 0; mt < 4; ++mt)
#pragma unroll
      for (int nt = 0; nt < 4; ++nt)
        acc[mt][nt] = WMMA_BF16(af[mt].v, bf[nt].v, acc[mt][nt]);
  }

#pragma unroll
  for (int mt = 0; mt < 4; ++mt)
#pragma unroll
    for (int nt = 0; nt < 4; ++nt) {
      const int n = colBase + nt * 16 + ln;
      const float bo = bias[n];
#pragma unroll
      for (int i = 0; i < 8; ++i) {
        const int m = rowBase + mt * 16 + half * 8 + i;
        Out[(size_t)m * C_ + n] = acc[mt][nt][i] + bo;
      }
    }
}

// ---------------------------------------------------------------- launch
extern "C" void kernel_launch(void* const* d_in, const int* in_sizes, int n_in,
                              void* d_out, int out_size, void* d_ws, size_t ws_size,
                              hipStream_t stream) {
  const float* x  = (const float*)d_in[0];
  const float* Wq = (const float*)d_in[1];
  const float* Wk = (const float*)d_in[2];
  const float* Wv = (const float*)d_in[3];
  const float* Wo = (const float*)d_in[4];
  const float* bo = (const float*)d_in[5];
  float* out = (float*)d_out;

  const size_t nX = (size_t)M_ * C_;   // 4,194,304
  const size_t nW = (size_t)C_ * C_;   //   262,144

  __bf16* xb   = (__bf16*)d_ws;
  __bf16* qb   = xb  + nX;
  __bf16* kb   = qb  + nX;
  __bf16* vtb  = kb  + nX;
  __bf16* attb = vtb + nX;
  __bf16* wqt  = attb + nX;
  __bf16* wkt  = wqt + nW;
  __bf16* wvt  = wkt + nW;
  __bf16* wot  = wvt + nW;

  // 1. convert inputs to bf16 (weights transposed)
  k_cvt<<<(int)(nX / 256), 256, 0, stream>>>(x, xb, (int)nX);
  k_cvtT<<<(int)(nW / 256), 256, 0, stream>>>(Wq, wqt);
  k_cvtT<<<(int)(nW / 256), 256, 0, stream>>>(Wk, wkt);
  k_cvtT<<<(int)(nW / 256), 256, 0, stream>>>(Wv, wvt);
  k_cvtT<<<(int)(nW / 256), 256, 0, stream>>>(Wo, wot);

  // 2. Q, K -> [B,H,L,DH]; V -> [B,H,DH,L]
  dim3 ggrid(M_ / 256, C_ / 64);
  k_gemm_qkv<<<ggrid, 128, 0, stream>>>(xb, wqt, qb, 1);
  k_gemm_qkv<<<ggrid, 128, 0, stream>>>(xb, wkt, kb, 1);
  k_gemm_qkv<<<ggrid, 128, 0, stream>>>(xb, wvt, vtb, 2);

  // 3. flash attention -> att [B,L,C] bf16
  k_attn<<<dim3(L_ / 128, B_ * H_), 128, 0, stream>>>(qb, kb, vtb, attb);

  // 4. out = att @ Wo + bo (f32)
  k_gemm_out<<<ggrid, 128, 0, stream>>>(attb, wot, bo, out);
}
